// MeasureSDPA_53180285059418
// MI455X (gfx1250) — matrix operand order repaired and checked by reading, not compiled, not run on hardware
//
#include <hip/hip_runtime.h>
#include <hip/hip_bf16.h>

#define BB 2
#define SS 2048
#define DMOD 1024
#define NH 16
#define DHEAD 64

typedef __bf16 bf16;
typedef bf16 v16bf __attribute__((ext_vector_type(16)));
typedef bf16 v8bf  __attribute__((ext_vector_type(8)));
typedef float v8f  __attribute__((ext_vector_type(8)));
typedef unsigned int v4u __attribute__((ext_vector_type(4)));
typedef int v8i __attribute__((ext_vector_type(8)));
typedef int v4i __attribute__((ext_vector_type(4)));

static __device__ __forceinline__ v8f wmma_bf16(v16bf a, v16bf b, v8f c) {
  // D = A(16x32 bf16) * B(32x16 bf16) + C(16x16 f32)
  return __builtin_amdgcn_wmma_f32_16x16x32_bf16(false, a, false, b, (short)0, c,
                                                 false, false);
}

// ---------------------------------------------------------------------------
// TDM 2-D tile load: global -> LDS.  D# packing per CDNA5 ISA
// group0: count=1 | lds_addr | global_addr[56:0] | type=2
// group1: data_size=2B | tensor_dim0/1 | tile_dim0/1 | tensor_dim0_stride
// ---------------------------------------------------------------------------
static __device__ __forceinline__ void tdm_load_2d(
    const void* gptr, unsigned lds_byte_off,
    unsigned tensor_d0, unsigned tensor_d1, unsigned d0_stride,
    unsigned tile_d0, unsigned tile_d1) {
  unsigned long long ga = (unsigned long long)gptr;
  v4u g0;
  g0.x = 1u;                                            // count=1, user mode
  g0.y = lds_byte_off;                                  // lds_addr (bytes)
  g0.z = (unsigned)(ga & 0xFFFFFFFFu);                  // global_addr[31:0]
  g0.w = (unsigned)((ga >> 32) & 0x1FFFFFFu) | (2u << 30);  // [56:32] | type=2
  v8i g1;
  g1[0] = (int)(1u << 16);                     // wg_mask=0, data_size=1 (2B)
  g1[1] = (int)(tensor_d0 << 16);              // tensor_dim0[15:0]
  g1[2] = (int)((tensor_d0 >> 16) | (tensor_d1 << 16));
  g1[3] = (int)((tensor_d1 >> 16) | (tile_d0 << 16));  // tile_dim0
  g1[4] = (int)(tile_d1 & 0xFFFFu);            // tile_dim1, tile_dim2=0
  g1[5] = (int)d0_stride;                      // tensor_dim0_stride[31:0]
  g1[6] = 0;
  g1[7] = 0;
  v4i z4 = {0, 0, 0, 0};
#if __has_include(<hip/amd_detail/amd_gfx1250_TDM.h>)
  v8i z8 = {0, 0, 0, 0, 0, 0, 0, 0};
  __builtin_amdgcn_tensor_load_to_lds(g0, g1, z4, z4, z8, 0);
#else
  __builtin_amdgcn_tensor_load_to_lds(g0, g1, z4, z4, 0);
#endif
}

// ---------------------------------------------------------------------------
// fp32 -> bf16 cast (one-time, amortized across heavy operand reuse)
// ---------------------------------------------------------------------------
__global__ void cast_f32_to_bf16(const float* __restrict__ in,
                                 bf16* __restrict__ out, int n) {
  int i = blockIdx.x * blockDim.x + threadIdx.x;
  int stride = gridDim.x * blockDim.x;
  for (; i < n; i += stride) out[i] = (bf16)in[i];
}

// ---------------------------------------------------------------------------
// QKV projection:  C[m,n] = sum_k X[m,k] * W[n,k] + bias[n]
// One wave computes a 16(M) x 64(N) output tile (A fragment reused 4x).
// ---------------------------------------------------------------------------
__global__ __launch_bounds__(256) void qkv_proj_wmma(
    const bf16* __restrict__ Xb,
    const bf16* __restrict__ Wqb, const bf16* __restrict__ Wkb,
    const bf16* __restrict__ Wvb,
    const float* __restrict__ bq, const float* __restrict__ bk,
    const float* __restrict__ bv,
    bf16* __restrict__ Qm, bf16* __restrict__ Km, bf16* __restrict__ Vt,
    float* __restrict__ scal) {
  const int p = blockIdx.y;  // 0=Q, 1=K, 2=V
  const bf16* W = (p == 0) ? Wqb : (p == 1) ? Wkb : Wvb;
  const float* bias = (p == 0) ? bq : (p == 1) ? bk : bv;

  const int lane = threadIdx.x & 31;
  const int col = lane & 15;
  const int halfw = lane >> 4;
  const int wave = blockIdx.x * (blockDim.x >> 5) + (threadIdx.x >> 5);  // 0..4095
  const int mtile = wave & 255;
  const int nt4 = wave >> 8;
  const int m0 = mtile * 16;
  const int n0 = nt4 * 64;

  v8f acc[4] = {};

  for (int kk = 0; kk < DMOD; kk += 32) {
    v16bf a;
    const bf16* Ap = Xb + (size_t)(m0 + col) * DMOD + kk;
    ((v8bf*)&a)[0] = *(const v8bf*)(Ap + halfw * 16);
    ((v8bf*)&a)[1] = *(const v8bf*)(Ap + 8 + halfw * 16);
#pragma unroll
    for (int t = 0; t < 4; ++t) {
      const bf16* Bp = W + (size_t)(n0 + t * 16 + col) * DMOD + kk + halfw * 16;
      v16bf bfr = *(const v16bf*)Bp;
      acc[t] = wmma_bf16(a, bfr, acc[t]);
    }
  }

  float amax = 0.f;
#pragma unroll
  for (int t = 0; t < 4; ++t) {
    const int nglob = n0 + t * 16 + col;
    const float bn = bias[nglob];
    const int h = nglob >> 6;
    const int d = nglob & (DHEAD - 1);
#pragma unroll
    for (int j = 0; j < 8; ++j) {
      const int m = m0 + j + 8 * halfw;
      const int bidx = m >> 11;
      const int s = m & (SS - 1);
      float v = acc[t][j] + bn;
      amax = fmaxf(amax, fabsf(v));
      if (p == 2) {
        Vt[((size_t)(bidx * NH + h) * DHEAD + d) * SS + s] = (bf16)v;
      } else {
        bf16* dst = (p == 0) ? Qm : Km;
        dst[((size_t)(bidx * NH + h) * SS + s) * DHEAD + d] = (bf16)v;
      }
    }
  }
  const int slot = (p == 0) ? 0 : (p == 1) ? 1 : 4;  // q_max, kT_max, v_max
  atomicMax((unsigned int*)(scal + slot), __float_as_uint(amax));
}

// ---------------------------------------------------------------------------
// Fused flash attention, TDM-fed:
//   block = 4 waves, each owns a 16-query tile (64 queries per WG).
//   K (32x64) and V^T (64x32) tiles double-buffered in LDS, filled by the
//   Tensor Data Mover (wave 0 issues, s_wait_tensorcnt, barrier releases).
//   Per 32-key chunk: 4 wmma (S) + online softmax + P bounce + 4 wmma (O).
// LDS map (bytes):  shK: [0, 8K)  shV: [8K, 16K)  shP: [16K, 20K)
// ---------------------------------------------------------------------------
__global__ __launch_bounds__(128) void flash_attn_wmma(
    const bf16* __restrict__ Qm, const bf16* __restrict__ Km,
    const bf16* __restrict__ Vt, float* __restrict__ out,
    float* __restrict__ scal) {
  __shared__ bf16 smem[2 * 2048 + 2 * 2048 + 4 * 512];  // 20 KB

  const int tid = threadIdx.x;
  const int wv = tid >> 5;
  const int lane = tid & 31;
  const int col = lane & 15;
  const int halfw = lane >> 4;
  const int q0 = (blockIdx.x * 4 + wv) * 16;
  const int h = blockIdx.y;
  const int b = blockIdx.z;
  const size_t bh = (size_t)(b * NH + h);

  const bf16* Kbase = Km + bh * SS * DHEAD;
  const bf16* Vbase = Vt + bh * DHEAD * SS;

  // Q tile as two A-fragments (d = 0..31 and 32..63), contiguous per lane
  const bf16* Qp = Qm + (bh * SS + q0 + col) * DHEAD;
  v16bf a0, a1;
  ((v8bf*)&a0)[0] = *(const v8bf*)(Qp + halfw * 16);
  ((v8bf*)&a0)[1] = *(const v8bf*)(Qp + 8 + halfw * 16);
  ((v8bf*)&a1)[0] = *(const v8bf*)(Qp + 32 + halfw * 16);
  ((v8bf*)&a1)[1] = *(const v8bf*)(Qp + 40 + halfw * 16);

  float mrow[8], lrow[8];
  v8f O[4] = {};
#pragma unroll
  for (int j = 0; j < 8; ++j) { mrow[j] = -1e30f; lrow[j] = 0.f; }
  float qkabs = 0.f;

  // preload chunk 0 into buffer 0 (TDM; EXEC-independent, wave 0 only)
  if (wv == 0) {
    // K tile: rows = 32 keys, 64 contiguous d each
    tdm_load_2d(Kbase, 0u, DHEAD, SS, DHEAD, DHEAD, 32);
    // V tile: rows = 64 d, 32 contiguous keys each (V stored transposed)
    tdm_load_2d(Vbase, 8192u, SS, DHEAD, SS, 32, DHEAD);
  }

  for (int k0 = 0; k0 < SS; k0 += 32) {
    const int buf = (k0 >> 5) & 1;
    if (wv == 0) {
      if (k0 + 32 < SS) {  // issue next chunk into the other buffer
        const unsigned nb = (unsigned)(buf ^ 1);
        tdm_load_2d(Kbase + (size_t)(k0 + 32) * DHEAD, nb * 4096u,
                    DHEAD, SS, DHEAD, DHEAD, 32);
        tdm_load_2d(Vbase + (size_t)(k0 + 32), 8192u + nb * 4096u,
                    SS, DHEAD, SS, 32, DHEAD);
        __builtin_amdgcn_s_wait_tensorcnt(2);  // current chunk landed
      } else {
        __builtin_amdgcn_s_wait_tensorcnt(0);
      }
    }
    __syncthreads();  // release compute waves; LDS tiles valid

    const bf16* shK = smem + buf * 2048;         // 32 x 64
    const bf16* shV = smem + 4096 + buf * 2048;  // 64 x 32
    bf16* shP = smem + 8192 + wv * 512;          // per-wave 16 x 32 bounce

    // --- S tile: 16 queries x 32 keys --------------------------------------
    v8f s0 = {}, s1 = {};
    {
      const bf16* Kp0 = shK + (size_t)col * DHEAD + halfw * 16;
      v16bf b00 = *(const v16bf*)(Kp0);
      v16bf b01 = *(const v16bf*)(Kp0 + 32);
      s0 = wmma_bf16(a0, b00, s0);
      s0 = wmma_bf16(a1, b01, s0);
      const bf16* Kp1 = Kp0 + 16 * DHEAD;
      v16bf b10 = *(const v16bf*)(Kp1);
      v16bf b11 = *(const v16bf*)(Kp1 + 32);
      s1 = wmma_bf16(a0, b10, s1);
      s1 = wmma_bf16(a1, b11, s1);
    }

    // --- online softmax (row j + 8*halfw lives in one 16-lane half) --------
    float e0[8], e1[8];
#pragma unroll
    for (int j = 0; j < 8; ++j) {
      float r0 = s0[j], r1 = s1[j];
      qkabs = fmaxf(qkabs, fmaxf(fabsf(r0), fabsf(r1)));  // pre-scale absmax
      r0 *= 0.125f;  // 1/sqrt(64)
      r1 *= 0.125f;
      float rm = fmaxf(r0, r1);
      rm = fmaxf(rm, __shfl_xor(rm, 1, 16));
      rm = fmaxf(rm, __shfl_xor(rm, 2, 16));
      rm = fmaxf(rm, __shfl_xor(rm, 4, 16));
      rm = fmaxf(rm, __shfl_xor(rm, 8, 16));
      float mnew = fmaxf(mrow[j], rm);
      float alpha = __expf(mrow[j] - mnew);
      mrow[j] = mnew;
      float p0 = __expf(r0 - mnew);
      float p1 = __expf(r1 - mnew);
      float rs = p0 + p1;
      rs += __shfl_xor(rs, 1, 16);
      rs += __shfl_xor(rs, 2, 16);
      rs += __shfl_xor(rs, 4, 16);
      rs += __shfl_xor(rs, 8, 16);
      lrow[j] = lrow[j] * alpha + rs;
      e0[j] = p0;
      e1[j] = p1;
#pragma unroll
      for (int t = 0; t < 4; ++t) O[t][j] *= alpha;
    }

    // --- relayer P: C-layout -> row-major LDS -> A-layout (same wave; DS
    //     ops are in-order per wave, so no barrier needed) ------------------
#pragma unroll
    for (int j = 0; j < 8; ++j) {
      shP[(j + 8 * halfw) * 32 + col] = (bf16)e0[j];
      shP[(j + 8 * halfw) * 32 + 16 + col] = (bf16)e1[j];
    }
    v16bf pa;
    ((v8bf*)&pa)[0] = *(const v8bf*)(shP + col * 32 + halfw * 16);
    ((v8bf*)&pa)[1] = *(const v8bf*)(shP + col * 32 + 8 + halfw * 16);

    // --- O += P(16x32) * V(32x64) from LDS ---------------------------------
#pragma unroll
    for (int t = 0; t < 4; ++t) {
      v16bf bvv = *(const v16bf*)(shV + (size_t)(t * 16 + col) * 32 + halfw * 16);
      O[t] = wmma_bf16(pa, bvv, O[t]);
    }
    __syncthreads();  // all waves done with buf before TDM overwrites it
  }

  // --- epilogue: normalize, write [B,S,H*DH] fp32, scalars -----------------
  float awmax = 0.f;
  float inv[8];
#pragma unroll
  for (int j = 0; j < 8; ++j) {
    inv[j] = 1.f / lrow[j];
    awmax = fmaxf(awmax, inv[j]);  // max prob per row = exp(0)/l = 1/l
  }
#pragma unroll
  for (int t = 0; t < 4; ++t) {
#pragma unroll
    for (int j = 0; j < 8; ++j) {
      const int q = q0 + j + 8 * halfw;
      const int cn = h * DHEAD + t * 16 + col;
      out[(size_t)(b * SS + q) * DMOD + cn] = O[t][j] * inv[j];
    }
  }
  atomicMax((unsigned int*)(scal + 2), __float_as_uint(qkabs));  // qk_out_max
  atomicMax((unsigned int*)(scal + 3), __float_as_uint(awmax));  // aw_max
  atomicMax((unsigned int*)(scal + 5), __float_as_uint(awmax));  // v_out_max
}

// ---------------------------------------------------------------------------
extern "C" void kernel_launch(void* const* d_in, const int* in_sizes, int n_in,
                              void* d_out, int out_size, void* d_ws,
                              size_t ws_size, hipStream_t stream) {
  (void)in_sizes; (void)n_in; (void)out_size; (void)ws_size;

  const float* hs = (const float*)d_in[0];
  const float* Wq = (const float*)d_in[1];
  const float* bq = (const float*)d_in[2];
  const float* Wk = (const float*)d_in[3];
  const float* bk = (const float*)d_in[4];
  const float* Wv = (const float*)d_in[5];
  const float* bv = (const float*)d_in[6];

  char* ws = (char*)d_ws;
  const size_t MB = (size_t)1 << 20;
  bf16* Xb  = (bf16*)(ws + 0);        //  8 MB: hidden bf16 [4096,1024]
  bf16* Wqb = (bf16*)(ws + 8 * MB);   //  2 MB
  bf16* Wkb = (bf16*)(ws + 10 * MB);  //  2 MB
  bf16* Wvb = (bf16*)(ws + 12 * MB);  //  2 MB
  bf16* Qm  = (bf16*)(ws + 14 * MB);  //  8 MB: [B,H,S,DH]
  bf16* Km  = (bf16*)(ws + 22 * MB);  //  8 MB: [B,H,S,DH]
  bf16* Vt  = (bf16*)(ws + 30 * MB);  //  8 MB: [B,H,DH,S] (transposed)

  float* out = (float*)d_out;
  float* scal = out + (size_t)BB * SS * DMOD;  // 6 scalar outputs

  hipMemsetAsync(scal, 0, 6 * sizeof(float), stream);

  const int nX = BB * SS * DMOD;
  const int nW = DMOD * DMOD;
  cast_f32_to_bf16<<<2048, 256, 0, stream>>>(hs, Xb, nX);
  cast_f32_to_bf16<<<1024, 256, 0, stream>>>(Wq, Wqb, nW);
  cast_f32_to_bf16<<<1024, 256, 0, stream>>>(Wk, Wkb, nW);
  cast_f32_to_bf16<<<1024, 256, 0, stream>>>(Wv, Wvb, nW);

  qkv_proj_wmma<<<dim3(512, 3), 256, 0, stream>>>(Xb, Wqb, Wkb, Wvb, bq, bk, bv,
                                                  Qm, Km, Vt, scal);

  // 4 query tiles per WG share TDM-staged K/V tiles
  flash_attn_wmma<<<dim3(SS / 64, NH, BB), 128, 0, stream>>>(Qm, Km, Vt, out,
                                                             scal);
}
